// AttentionHead_50500225466666
// MI455X (gfx1250) — compile-verified, matrix operand order checked by reading, not run on hardware
//
#include <hip/hip_runtime.h>

#define D_MODEL 1024
#define D_VALUE 64
#define N_CTX   8192

typedef __attribute__((ext_vector_type(16))) __bf16 v16bf;
typedef __attribute__((ext_vector_type(8)))  __bf16 v8bf;
typedef __attribute__((ext_vector_type(8)))  float  v8f;

static __device__ __forceinline__ v16bf cat8(v8bf lo, v8bf hi) {
  return __builtin_shufflevector(lo, hi, 0,1,2,3,4,5,6,7,8,9,10,11,12,13,14,15);
}

static __device__ __forceinline__ v8f wmma_bf16(v16bf a, v16bf b, v8f c) {
  // D = A(16x32 bf16) * B(32x16 bf16) + C(16x16 f32)
  return __builtin_amdgcn_wmma_f32_16x16x32_bf16(false, a, false, b,
                                                 (short)0, c, false, false);
}

// Loads one 16x32-bf16 A/B fragment: two contiguous 16B chunks per lane.
static __device__ __forceinline__ v16bf frag_ld(const __bf16* p) {
  return cat8(*(const v8bf*)(p), *(const v8bf*)(p + 16));
}

// ---------------------------------------------------------------------------
// Kernel 1: QKV projection.  out[t][j] (or out[j][t] transposed for V) =
//   sum_d x[d][t] * W[j][d] + b[j], stored as bf16.
// x reads are coalesced over t; W/b indices are wave-uniform -> scalar loads.
// ---------------------------------------------------------------------------
__global__ __launch_bounds__(256) void qkv_proj_kernel(
    const float* __restrict__ x, const float* __restrict__ W,
    const float* __restrict__ b, __bf16* __restrict__ out, int transposed) {
  const int t = blockIdx.x * blockDim.x + threadIdx.x;
  float acc[D_VALUE];
#pragma unroll
  for (int j = 0; j < D_VALUE; ++j) acc[j] = 0.0f;
  for (int d = 0; d < D_MODEL; ++d) {
    const float xv = x[d * N_CTX + t];
#pragma unroll
    for (int j = 0; j < D_VALUE; ++j) acc[j] += xv * W[j * D_MODEL + d];
  }
  if (!transposed) {
#pragma unroll
    for (int j = 0; j < D_VALUE; ++j)
      out[t * D_VALUE + j] = (__bf16)(acc[j] + b[j]);
  } else {
#pragma unroll
    for (int j = 0; j < D_VALUE; ++j)
      out[j * N_CTX + t] = (__bf16)(acc[j] + b[j]);
  }
}

// ---------------------------------------------------------------------------
// Kernel 2: causal flash attention.  Per wave: one 16-row Q tile, iterate KV
// in 64-row tiles.  S = Q K^T via 8x v_wmma_f32_16x16x32_bf16, online softmax
// in the C-layout (row stats via xor-shuffles inside 16-lane halves), P is
// routed through LDS into the A-layout, O += P V via 8 more WMMAs against vT.
// All control flow is SGPR-based (t0 forced uniform) so EXEC stays all-ones.
// Writes accT[d][t] = (attn @ v)^T in f32.
// ---------------------------------------------------------------------------
#define P_STRIDE 72  // bf16 elems per LDS row (64 + pad; 144B keeps 16B align)

__global__ __launch_bounds__(128) void flash_attn_kernel(
    const __bf16* __restrict__ q, const __bf16* __restrict__ k,
    const __bf16* __restrict__ vT, float* __restrict__ accT) {
  __shared__ __align__(16) __bf16 Plds[4][16 * P_STRIDE];

  const int wid  = threadIdx.x >> 5;
  const int lane = threadIdx.x & 31;
  const int n16  = lane & 15;   // col within 16x16 tile / A-row index
  const int kh   = lane >> 4;   // half-wave select for K-dim packing
  // Force wave-uniform: loop bounds & mask branches become scalar branches.
  const int t0   = __builtin_amdgcn_readfirstlane((blockIdx.x * 4 + wid) * 16);
  __bf16* Pw = &Plds[wid][0];

  const float S2 = 0.125f * 1.44269504088896f;  // 1/sqrt(64) * log2(e)
  const float NEG_BIG = -3.0e38f;

  // Q A-fragments (row m = n16), loaded once.
  const __bf16* qrow = q + (size_t)(t0 + n16) * D_VALUE;
  const v16bf aQ0 = frag_ld(qrow + 0  + kh * 8);
  const v16bf aQ1 = frag_ld(qrow + 32 + kh * 8);

  float mi[8], li[8];
#pragma unroll
  for (int r = 0; r < 8; ++r) { mi[r] = NEG_BIG; li[r] = 0.0f; }
  v8f O0 = {}, O1 = {}, O2 = {}, O3 = {};

  for (int j0 = 0; j0 <= t0 + 15; j0 += 64) {
    // Prefetch next KV tiles (speculative; emits global_prefetch_b8).
    __builtin_prefetch((const void*)(k + (size_t)(j0 + 64 + n16) * D_VALUE), 0, 1);
    __builtin_prefetch((const void*)(vT + (size_t)n16 * N_CTX + j0 + 64), 0, 1);

    // ---- S = Q K^T : four 16x16 column tiles, K-dim 64 = 2 WMMAs each ----
    v8f c0 = {}, c1 = {}, c2 = {}, c3 = {};
    {
      const __bf16* kr0 = k + (size_t)(j0 + 0  + n16) * D_VALUE + kh * 8;
      const __bf16* kr1 = k + (size_t)(j0 + 16 + n16) * D_VALUE + kh * 8;
      const __bf16* kr2 = k + (size_t)(j0 + 32 + n16) * D_VALUE + kh * 8;
      const __bf16* kr3 = k + (size_t)(j0 + 48 + n16) * D_VALUE + kh * 8;
      c0 = wmma_bf16(aQ0, frag_ld(kr0), c0);
      c0 = wmma_bf16(aQ1, frag_ld(kr0 + 32), c0);
      c1 = wmma_bf16(aQ0, frag_ld(kr1), c1);
      c1 = wmma_bf16(aQ1, frag_ld(kr1 + 32), c1);
      c2 = wmma_bf16(aQ0, frag_ld(kr2), c2);
      c2 = wmma_bf16(aQ1, frag_ld(kr2 + 32), c2);
      c3 = wmma_bf16(aQ0, frag_ld(kr3), c3);
      c3 = wmma_bf16(aQ1, frag_ld(kr3 + 32), c3);
    }

    // ---- causal mask (scalar branch: j0, t0 are SGPRs; EXEC untouched) ----
    if (j0 + 63 > t0) {
#pragma unroll
      for (int r = 0; r < 8; ++r) {
        const int trow = t0 + r + 8 * kh;
        c0[r] = (j0 + 0  + n16 > trow) ? NEG_BIG : c0[r];
        c1[r] = (j0 + 16 + n16 > trow) ? NEG_BIG : c1[r];
        c2[r] = (j0 + 32 + n16 > trow) ? NEG_BIG : c2[r];
        c3[r] = (j0 + 48 + n16 > trow) ? NEG_BIG : c3[r];
      }
    }

    // ---- online softmax update (shuffles amortized over 64 KV cols) ----
#pragma unroll
    for (int r = 0; r < 8; ++r) {
      float rm = fmaxf(fmaxf(c0[r], c1[r]), fmaxf(c2[r], c3[r]));
      rm = fmaxf(rm, __shfl_xor(rm, 1, 32));
      rm = fmaxf(rm, __shfl_xor(rm, 2, 32));
      rm = fmaxf(rm, __shfl_xor(rm, 4, 32));
      rm = fmaxf(rm, __shfl_xor(rm, 8, 32));
      const float mnew  = fmaxf(mi[r], rm);
      const float alpha = exp2f((mi[r] - mnew) * S2);
      const float p0 = exp2f((c0[r] - mnew) * S2);
      const float p1 = exp2f((c1[r] - mnew) * S2);
      const float p2 = exp2f((c2[r] - mnew) * S2);
      const float p3 = exp2f((c3[r] - mnew) * S2);
      float rs = (p0 + p1) + (p2 + p3);
      rs += __shfl_xor(rs, 1, 32);
      rs += __shfl_xor(rs, 2, 32);
      rs += __shfl_xor(rs, 4, 32);
      rs += __shfl_xor(rs, 8, 32);
      li[r] = li[r] * alpha + rs;
      mi[r] = mnew;
      O0[r] *= alpha; O1[r] *= alpha; O2[r] *= alpha; O3[r] *= alpha;
      // Stash P in LDS (C-layout -> row-major [16][64])
      const int row = r + 8 * kh;
      Pw[row * P_STRIDE + 0  + n16] = (__bf16)p0;
      Pw[row * P_STRIDE + 16 + n16] = (__bf16)p1;
      Pw[row * P_STRIDE + 32 + n16] = (__bf16)p2;
      Pw[row * P_STRIDE + 48 + n16] = (__bf16)p3;
    }

    // ---- P A-fragments from LDS (ds ops are in-order within a wave) ----
    const __bf16* prow = Pw + n16 * P_STRIDE + kh * 8;
    const v16bf aP0 = frag_ld(prow);
    const v16bf aP1 = frag_ld(prow + 32);

    // ---- O += P V : B-fragments from vT, contiguous 16B chunks ----
    const __bf16* vr = vT + (size_t)n16 * N_CTX + j0 + kh * 8;
    O0 = wmma_bf16(aP0, frag_ld(vr + 0 * 16 * N_CTX), O0);
    O0 = wmma_bf16(aP1, frag_ld(vr + 0 * 16 * N_CTX + 32), O0);
    O1 = wmma_bf16(aP0, frag_ld(vr + 1 * 16 * N_CTX), O1);
    O1 = wmma_bf16(aP1, frag_ld(vr + 1 * 16 * N_CTX + 32), O1);
    O2 = wmma_bf16(aP0, frag_ld(vr + 2 * 16 * N_CTX), O2);
    O2 = wmma_bf16(aP1, frag_ld(vr + 2 * 16 * N_CTX + 32), O2);
    O3 = wmma_bf16(aP0, frag_ld(vr + 3 * 16 * N_CTX), O3);
    O3 = wmma_bf16(aP1, frag_ld(vr + 3 * 16 * N_CTX + 32), O3);
  }

  // ---- epilogue: accT[d][t] = O / l ----
#pragma unroll
  for (int r = 0; r < 8; ++r) {
    const float inv = 1.0f / li[r];
    const int t = t0 + r + 8 * kh;
    accT[(size_t)(0  + n16) * N_CTX + t] = O0[r] * inv;
    accT[(size_t)(16 + n16) * N_CTX + t] = O1[r] * inv;
    accT[(size_t)(32 + n16) * N_CTX + t] = O2[r] * inv;
    accT[(size_t)(48 + n16) * N_CTX + t] = O3[r] * inv;
  }
}

// ---------------------------------------------------------------------------
// Kernel 3: out[d][t] = W_O[d][:] . accT[:][t] + b_O[d]   (f32)
// ---------------------------------------------------------------------------
__global__ __launch_bounds__(256) void out_proj_kernel(
    const float* __restrict__ accT, const float* __restrict__ WO,
    const float* __restrict__ bO, float* __restrict__ out) {
  const int t = blockIdx.x * blockDim.x + threadIdx.x;
  float a[D_VALUE];
#pragma unroll
  for (int j = 0; j < D_VALUE; ++j) a[j] = accT[(size_t)j * N_CTX + t];
  for (int d = 0; d < D_MODEL; ++d) {
    float s = bO[d];
#pragma unroll
    for (int j = 0; j < D_VALUE; ++j) s += WO[d * D_VALUE + j] * a[j];
    out[(size_t)d * N_CTX + t] = s;
  }
}

// ---------------------------------------------------------------------------
extern "C" void kernel_launch(void* const* d_in, const int* in_sizes, int n_in,
                              void* d_out, int out_size, void* d_ws,
                              size_t ws_size, hipStream_t stream) {
  const float* x  = (const float*)d_in[0];
  const float* WQ = (const float*)d_in[1];
  const float* bQ = (const float*)d_in[2];
  const float* WK = (const float*)d_in[3];
  const float* bK = (const float*)d_in[4];
  const float* WV = (const float*)d_in[5];
  const float* bV = (const float*)d_in[6];
  const float* WO = (const float*)d_in[7];
  const float* bO = (const float*)d_in[8];
  float* out = (float*)d_out;

  char* ws = (char*)d_ws;
  const size_t QKV_BYTES = (size_t)N_CTX * D_VALUE * sizeof(__bf16);  // 1 MB
  __bf16* qb   = (__bf16*)(ws);
  __bf16* kb   = (__bf16*)(ws + QKV_BYTES);
  __bf16* vTb  = (__bf16*)(ws + 2 * QKV_BYTES);
  float*  accT = (float*)(ws + 3 * QKV_BYTES);  // 64*8192 f32 = 2 MB

  qkv_proj_kernel<<<dim3(N_CTX / 256), 256, 0, stream>>>(x, WQ, bQ, qb, 0);
  qkv_proj_kernel<<<dim3(N_CTX / 256), 256, 0, stream>>>(x, WK, bK, kb, 0);
  qkv_proj_kernel<<<dim3(N_CTX / 256), 256, 0, stream>>>(x, WV, bV, vTb, 1);
  flash_attn_kernel<<<dim3(N_CTX / 16 / 4), 128, 0, stream>>>(qb, kb, vTb, accT);
  out_proj_kernel<<<dim3(N_CTX / 256), 256, 0, stream>>>(accT, WO, bO, out);
}
